// self_attention_85040352461375
// MI455X (gfx1250) — matrix operand order repaired and checked by reading, not compiled, not run on hardware
//
#include <hip/hip_runtime.h>
#include <math.h>

// ---------------------------------------------------------------------------
// Self-attention (softmax over output index m) for MI455X gfx1250.
// B=8, C_IN=12, C_QKV=36, N=4096.
// Heavy math: v_wmma_f32_16x16x32_bf16.  Data movement: double-buffered
// global_load_async_to_lds_b128 shared by all 8 waves of a block.
// Softmax uses the exact unnormalized form exp(s)/sum(exp(s)) (safe: |s|<<88
// for this data), and log2(e) is folded into the K scale so the exponential
// is a single native v_exp_f32 (exp2) per element.
// ---------------------------------------------------------------------------

typedef __attribute__((ext_vector_type(16))) __bf16 v16bf;
typedef __attribute__((ext_vector_type(2)))  __bf16 bf16x2;
typedef __attribute__((ext_vector_type(8)))  float  v8f;

#define B_    8
#define CIN   12
#define CQKV  36
#define N_    4096
#define NT    (N_ / 16)          // 256 tiles of 16 along n / m
#define NS32  (N_ / 32)          // 128 steps of 32 along n
// (1/sqrt(12)) * log2(e): softmax scale and exp->exp2 conversion, pre-folded
#define SCALE_L2E (0.28867513459481287f * 1.4426950408889634f)

// Workspace layout (byte offsets into d_ws):
//   qB  : bf16 [B][NT][2][512]   B-layout Q tiles (c padded to 64)     4 MB
//   ktA : bf16 [B][NT][2][512]   A-layout K^T tiles, scale*log2e folded 4 MB
//   vA  : bf16 [B][3][NS32][512] A-layout V tiles (c padded to 48)     3 MB
//   st  : float [B][N]           1/sum_m exp2(s[n,m])                128 KB
#define OFF_QB   0
#define OFF_KTA  4194304
#define OFF_VA   8388608
#define OFF_ST   11534336
// total required ws: 11,665,408 bytes

static __device__ __forceinline__ v8f wmma_bf16(v16bf a, v16bf b, v8f c) {
  return __builtin_amdgcn_wmma_f32_16x16x32_bf16(false, a, false, b,
                                                 (short)0, c, false, false);
}

// CDNA5 async copy global -> LDS (ASYNCcnt tracked), 16B per active lane.
// lds_off is an absolute LDS byte offset (dynamic-LDS base comes from
// __builtin_amdgcn_groupstaticsize()).
static __device__ __forceinline__ void async_ld_b128(unsigned lds_off,
                                                     unsigned long long gaddr) {
  asm volatile("global_load_async_to_lds_b128 %0, %1, off"
               :: "v"(lds_off), "v"(gaddr)
               : "memory");
}
static __device__ __forceinline__ void wait_async0() {
  asm volatile("s_wait_asynccnt 0x0" ::: "memory");
}

// ---------------------------------------------------------------------------
// Kernel 1: QKV projections + pack into WMMA-native bf16 layouts.
// One thread per (b, n). 16-bit A-layout (16x32): lane = (K/16)*16 + M,
// elem = K%16.  16-bit B-layout (32x16): lane = (K/16)*16+N,
// elem = K%16.  (contiguous k in both, chosen here; AMD's table order differs)
// ---------------------------------------------------------------------------
__global__ __launch_bounds__(256) void qkv_pack_kernel(
    const float* __restrict__ x,
    const float* __restrict__ wq, const float* __restrict__ bq,
    const float* __restrict__ wk, const float* __restrict__ bk,
    const float* __restrict__ wv, const float* __restrict__ bv,
    __bf16* __restrict__ qB, __bf16* __restrict__ ktA,
    __bf16* __restrict__ vA) {
  int idx = blockIdx.x * blockDim.x + threadIdx.x;   // 0 .. B*N-1
  int b = idx >> 12;
  int n = idx & (N_ - 1);

  float xv[CIN];
#pragma unroll
  for (int i = 0; i < CIN; ++i) xv[i] = x[((size_t)b * CIN + i) * N_ + n];

  for (int c = 0; c < 64; ++c) {
    float qv = 0.f, kv = 0.f, vv = 0.f;
    if (c < CQKV) {
      qv = bq[c]; kv = bk[c]; vv = bv[c];
#pragma unroll
      for (int i = 0; i < CIN; ++i) {
        qv += wq[c * CIN + i] * xv[i];
        kv += wk[c * CIN + i] * xv[i];
        vv += wv[c * CIN + i] * xv[i];
      }
    }
    int kk  = c & 31;
    int blk = c >> 5;
    size_t tbase = ((((size_t)b * NT + (n >> 4)) * 2 + blk) << 9);
    // Q as B-matrix operand (K = c, N = m = n)
    qB[tbase + ((((kk >> 4) << 4) + (n & 15)) << 4) + (kk & 15)] = (__bf16)qv;
    // K^T as A-matrix operand (M = n, K = c); fold scale * log2(e) into K so
    // scores come out of the WMMA in the exp2 domain.
    ktA[tbase + ((((kk >> 4) << 4) + (n & 15)) << 4) +
        (kk & 15)] = (__bf16)(kv * SCALE_L2E);
    // V as A-matrix operand (M = c, K = n), c padded to 48
    if (c < 48) {
      int nn = n & 31;
      size_t va = ((((size_t)b * 3 + (c >> 4)) * NS32 + (n >> 5)) << 9) +
                  ((((nn >> 4) << 4) + (c & 15)) << 4) +
                  (nn & 15);
      vA[va] = (__bf16)vv;
    }
  }
}

// ---------------------------------------------------------------------------
// Kernel 2: per-row denominator sum_m exp2(s[n,m]).  One wave per 16-row
// n-strip; all 8 waves of the block share the Q tile, streamed into LDS via
// double-buffered async copies.  C/D layout: VGPR r = row r + 8*(lane/16),
// col = lane%16 -> per-row reduction is a 16-lane-half butterfly sum.
// Dynamic LDS: 2 x 2048 B Q tile buffers.
// ---------------------------------------------------------------------------
__global__ __launch_bounds__(256) void rowstats_kernel(
    const __bf16* __restrict__ ktA, const __bf16* __restrict__ qB,
    float* __restrict__ stats) {
  extern __shared__ char smem[];
  const unsigned lbase = __builtin_amdgcn_groupstaticsize();
  int t = threadIdx.x;
  int lane = t & 31;
  int w    = t >> 5;
  int strip = blockIdx.x * 8 + w;     // 0 .. B*NT-1 (whole block: same b)
  int b  = strip >> 8;
  int nt = strip & 255;

  const char* qBb = (const char*)qB;

  // K^T strip for this wave: resident in registers for the whole pass
  const v16bf* Ap = (const v16bf*)(ktA + ((((size_t)b * NT + nt) * 2) << 9));
  v16bf a0 = Ap[lane];
  v16bf a1 = Ap[32 + lane];

  float rsum[8];
#pragma unroll
  for (int r = 0; r < 8; ++r) rsum[r] = 0.f;

  // prologue: stage Q tile for mt=0
  if (t < 128)
    async_ld_b128(lbase + t * 16,
                  (unsigned long long)(uintptr_t)(qBb + (size_t)(b * NT) * 2048 +
                                                  t * 16));
  wait_async0();
  __syncthreads();

  for (int mt = 0; mt < NT; ++mt) {
    int cur = mt & 1;
    if (mt + 1 < NT && t < 128)   // overlap next tile's copy with compute
      async_ld_b128(lbase + (cur ^ 1) * 2048 + t * 16,
                    (unsigned long long)(uintptr_t)(
                        qBb + (size_t)(b * NT + mt + 1) * 2048 + t * 16));
    const char* qb = smem + cur * 2048;
    v16bf b0 = *(const v16bf*)(qb + lane * 32);
    v16bf b1 = *(const v16bf*)(qb + 1024 + lane * 32);
    v8f s = {};
    s = wmma_bf16(a0, b0, s);
    s = wmma_bf16(a1, b1, s);
#pragma unroll
    for (int r = 0; r < 8; ++r)
      rsum[r] += __builtin_amdgcn_exp2f(s[r]);  // native v_exp_f32, no prescale
    wait_async0();
    __syncthreads();
  }
  // sum across the 16 lanes sharing each row (xor masks stay in-half)
#pragma unroll
  for (int mask = 1; mask <= 8; mask <<= 1)
#pragma unroll
    for (int r = 0; r < 8; ++r) rsum[r] += __shfl_xor(rsum[r], mask, 32);

  if ((lane & 15) == 0) {
#pragma unroll
    for (int r = 0; r < 8; ++r) {
      int n = (nt << 4) + r + ((lane >> 4) << 3);
      stats[(size_t)b * N_ + n] = 1.0f / rsum[r];
    }
  }
}

// ---------------------------------------------------------------------------
// Kernel 3: fused attention.  One wave per (b, 16-wide m-tile); all 8 waves
// of a block share (b, ns) so K^T/V tiles AND the 32 softmax denominators are
// streamed once per block into double-buffered LDS via async copies.
// Per 32-deep n-step: 4 WMMAs rebuild S, exp2 -> bf16 P staged in wave-private
// LDS already in B-layout, 3 WMMAs accumulate attn[c,m].  Epilogue applies wo.
// Dynamic LDS: K 2x4096 | V 2x3072 | inv 2x128 | P 8x1024 | attn 8x3072.
// ---------------------------------------------------------------------------
#define R_KBUF 0
#define R_VBUF 8192
#define R_SBUF 14336
#define R_PBUF 14592
#define R_ABUF 22784
#define SMEM_MAIN 47360

__global__ __launch_bounds__(256) void attn_main_kernel(
    const __bf16* __restrict__ ktA, const __bf16* __restrict__ qB,
    const __bf16* __restrict__ vA, const float* __restrict__ stats,
    const float* __restrict__ wo, const float* __restrict__ bo,
    float* __restrict__ out) {
  extern __shared__ char smem[];
  const unsigned lbase = __builtin_amdgcn_groupstaticsize();
  int t = threadIdx.x;
  int lane = t & 31;
  int w    = t >> 5;
  int tile = blockIdx.x * 8 + w;    // 0 .. B*NT-1 (whole block: same b)
  int b  = tile >> 8;
  int mt = tile & 255;

  const char* ktAb = (const char*)ktA;
  const char* vAb  = (const char*)vA;
  const char* stbb = (const char*)(stats + (size_t)b * N_);

  // prologue: stage K^T + V tiles + denominators for ns=0
  {
    async_ld_b128(lbase + R_KBUF + t * 16,
                  (unsigned long long)(uintptr_t)(
                      ktAb + (size_t)(b * NT) * 2048 + t * 16));
    if (t < 192) {
      int ct = t >> 6, o = (t & 63) * 16;
      async_ld_b128(lbase + R_VBUF + ct * 1024 + o,
                    (unsigned long long)(uintptr_t)(
                        vAb + (size_t)((b * 3 + ct) * NS32) * 1024 + o));
    }
    if (t < 8)
      async_ld_b128(lbase + R_SBUF + t * 16,
                    (unsigned long long)(uintptr_t)(stbb + t * 16));
  }

  // Q tile for this wave's m-tile: registers for the whole pass
  const v16bf* Qp = (const v16bf*)(qB + ((((size_t)b * NT + mt) * 2) << 9));
  v16bf bq0 = Qp[lane];
  v16bf bq1 = Qp[32 + lane];

  v8f acc0 = {}, acc1 = {}, acc2 = {};
  __bf16* pP = (__bf16*)(smem + R_PBUF + w * 1024);
  int m_l = lane & 15;
  int hi  = (lane >> 4) << 3;       // 0 or 8: row offset in C/D layout

  wait_async0();
  __syncthreads();

  for (int ns = 0; ns < NS32; ++ns) {
    int cur = ns & 1;
    if (ns + 1 < NS32) {            // overlap next step's copies with compute
      async_ld_b128(lbase + R_KBUF + (cur ^ 1) * 4096 + t * 16,
                    (unsigned long long)(uintptr_t)(
                        ktAb + (size_t)(b * NT + 2 * (ns + 1)) * 2048 + t * 16));
      if (t < 192) {
        int ct = t >> 6, o = (t & 63) * 16;
        async_ld_b128(lbase + R_VBUF + (cur ^ 1) * 3072 + ct * 1024 + o,
                      (unsigned long long)(uintptr_t)(
                          vAb + (size_t)((b * 3 + ct) * NS32 + ns + 1) * 1024 + o));
      }
      if (t < 8)
        async_ld_b128(lbase + R_SBUF + (cur ^ 1) * 128 + t * 16,
                      (unsigned long long)(uintptr_t)(
                          stbb + (size_t)(ns + 1) * 128 + t * 16));
    }
    const char* kb = smem + R_KBUF + cur * 4096;
    const char* vb = smem + R_VBUF + cur * 3072;
    const char* sb = smem + R_SBUF + cur * 128;
#pragma unroll
    for (int sub = 0; sub < 2; ++sub) {
      v16bf a0 = *(const v16bf*)(kb + sub * 2048 + lane * 32);
      v16bf a1 = *(const v16bf*)(kb + sub * 2048 + 1024 + lane * 32);
      v8f s = {};
      s = wmma_bf16(a0, bq0, s);
      s = wmma_bf16(a1, bq1, s);
      float inv8[8];
      *(float4*)&inv8[0] = *(const float4*)(sb + (sub * 16 + hi) * 4);
      *(float4*)&inv8[4] = *(const float4*)(sb + (sub * 16 + hi) * 4 + 16);
#pragma unroll
      for (int r = 0; r < 8; r += 2) {   // paired -> 4B ds stores / cvt_pk
        float p0 = __builtin_amdgcn_exp2f(s[r]) * inv8[r];
        float p1 = __builtin_amdgcn_exp2f(s[r + 1]) * inv8[r + 1];
        bf16x2 pk = {(__bf16)p0, (__bf16)p1};
        // B-layout: lane' = sub*16 + m_l, elem = row_local
        *(bf16x2*)(pP + ((sub << 4) + m_l) * 16 + (r + hi)) = pk;
      }
    }
    // P is wave-private; per-wave LDS ops are in-order -> scheduling fence only
    __builtin_amdgcn_wave_barrier();
    v16bf pB = *(const v16bf*)(pP + lane * 16);
    v16bf v0 = *(const v16bf*)(vb + lane * 32);
    v16bf v1 = *(const v16bf*)(vb + 1024 + lane * 32);
    v16bf v2 = *(const v16bf*)(vb + 2048 + lane * 32);
    acc0 = wmma_bf16(v0, pB, acc0);
    acc1 = wmma_bf16(v1, pB, acc1);
    acc2 = wmma_bf16(v2, pB, acc2);
    wait_async0();
    __syncthreads();
  }

  // stage attn tile (rows c=0..47, cols m=0..15) in LDS, then project with wo
  float* aL = (float*)(smem + R_ABUF + w * 3072);
#pragma unroll
  for (int r = 0; r < 8; ++r) {
    aL[(r + hi) * 16 + m_l]      = acc0[r];
    aL[(16 + r + hi) * 16 + m_l] = acc1[r];
    aL[(32 + r + hi) * 16 + m_l] = acc2[r];
  }
  __builtin_amdgcn_wave_barrier();
  int obase = (lane >> 4) * 6;      // lanes 0-15: o=0..5, lanes 16-31: o=6..11
#pragma unroll
  for (int oo = 0; oo < 6; ++oo) {
    int o = obase + oo;
    float sum = bo[o];
#pragma unroll
    for (int c = 0; c < CQKV; ++c) sum += wo[o * CQKV + c] * aL[c * 16 + m_l];
    out[((size_t)b * CIN + o) * N_ + (mt * 16 + m_l)] = sum;
  }
}

// ---------------------------------------------------------------------------
extern "C" void kernel_launch(void* const* d_in, const int* in_sizes, int n_in,
                              void* d_out, int out_size, void* d_ws,
                              size_t ws_size, hipStream_t stream) {
  const float* x  = (const float*)d_in[0];
  const float* wq = (const float*)d_in[1];
  const float* bq = (const float*)d_in[2];
  const float* wk = (const float*)d_in[3];
  const float* bk = (const float*)d_in[4];
  const float* wv = (const float*)d_in[5];
  const float* bv = (const float*)d_in[6];
  const float* wo = (const float*)d_in[7];
  const float* bo = (const float*)d_in[8];
  float* out = (float*)d_out;

  char* ws = (char*)d_ws;
  __bf16* qB   = (__bf16*)(ws + OFF_QB);
  __bf16* ktA  = (__bf16*)(ws + OFF_KTA);
  __bf16* vA   = (__bf16*)(ws + OFF_VA);
  float* stats = (float*)(ws + OFF_ST);

  // 1) project + pack QKV into WMMA-native bf16 layouts (fully rewritten
  //    every call, including zero padding -> deterministic)
  qkv_pack_kernel<<<(B_ * N_) / 256, 256, 0, stream>>>(
      x, wq, bq, wk, bk, wv, bv, qB, ktA, vA);
  // 2) per-row softmax denominators (1/sum exp2) via WMMA + async LDS Q tiles
  rowstats_kernel<<<(B_ * NT) / 8, 256, 4096, stream>>>(ktA, qB, stats);
  // 3) fused S-recompute + softmax-apply + V matmul + output projection
  attn_main_kernel<<<(B_ * NT) / 8, 256, SMEM_MAIN, stream>>>(
      ktA, qB, vA, stats, wo, bo, out);
}
